// MHA_73667279061622
// MI455X (gfx1250) — matrix operand order repaired and checked by reading, not compiled, not run on hardware
//
#include <hip/hip_runtime.h>
#include <hip/hip_bf16.h>

// ---------------------------------------------------------------------------
// MHA forward for MI455X (gfx1250), wave32 + WMMA bf16 path.
// B=8, S=1024, D=1024, H=16, HD=64.
// ---------------------------------------------------------------------------

typedef __attribute__((ext_vector_type(16))) __bf16 v16bf;
typedef __attribute__((ext_vector_type(8)))  __bf16 v8bf;
typedef __attribute__((ext_vector_type(8)))  float  v8f;
typedef __attribute__((ext_vector_type(4)))  float  f32x4;

#define NB  8
#define NS  1024
#define ND  1024
#define NH  16
#define NHD 64

__device__ __forceinline__ v8f wmma_bf16(v16bf a, v16bf b, v8f c) {
  // (neg_a, A, neg_b, B, c_mod, C, reuse_a, reuse_b)
  return __builtin_amdgcn_wmma_f32_16x16x32_bf16(false, a, false, b,
                                                 (short)0, c, false, false);
}

// A fragment (16x32 bf16) from an f32 row-major row pointer, optional scale.
// Lane half=lane>>4: elements 0..15 = K {half*16..+15} (contiguous).
__device__ __forceinline__ v16bf load_a_f32(const float* __restrict__ row,
                                            int kbase, int lane, float scale) {
  const int half = lane >> 4;
  const float* p0 = row + kbase + half * 16;
  f32x4 a0 = *(const f32x4*)(p0);
  f32x4 a1 = *(const f32x4*)(p0 + 4);
  f32x4 b0 = *(const f32x4*)(p0 + 8);
  f32x4 b1 = *(const f32x4*)(p0 + 12);
  v16bf r;
#pragma unroll
  for (int i = 0; i < 4; ++i) {
    r[i]      = (__bf16)(a0[i] * scale);
    r[4 + i]  = (__bf16)(a1[i] * scale);
    r[8 + i]  = (__bf16)(b0[i] * scale);
    r[12 + i] = (__bf16)(b1[i] * scale);
  }
  return r;
}

// A fragment from a bf16 row-major row pointer.
__device__ __forceinline__ v16bf load_a_bf16(const __bf16* __restrict__ row,
                                             int kbase, int lane) {
  const int half = lane >> 4;
  const __bf16* p0 = row + kbase + half * 16;
  v8bf lo = *(const v8bf*)(p0);
  v8bf hi = *(const v8bf*)(p0 + 8);
  v16bf r;
#pragma unroll
  for (int i = 0; i < 8; ++i) { r[i] = lo[i]; r[8 + i] = hi[i]; }
  return r;
}

// B fragment (32x16 bf16): lane = column; col pointer is K-contiguous.
// Lane half=lane>>4: 16 contiguous K at kbase + half*16  (one 32B load).
__device__ __forceinline__ v16bf load_b_bf16(const __bf16* __restrict__ col,
                                             int kbase, int lane) {
  const int half = lane >> 4;
  return *(const v16bf*)(col + kbase + half * 16);
}

// B fragment from f32 weight row (W row n is K-contiguous for Y = X*W^T).
__device__ __forceinline__ v16bf load_b_f32(const float* __restrict__ col,
                                            int kbase, int lane) {
  const int half = lane >> 4;
  const float* p = col + kbase + half * 16;
  f32x4 c0 = *(const f32x4*)(p);
  f32x4 c1 = *(const f32x4*)(p + 4);
  f32x4 c2 = *(const f32x4*)(p + 8);
  f32x4 c3 = *(const f32x4*)(p + 12);
  v16bf r;
#pragma unroll
  for (int i = 0; i < 4; ++i) {
    r[i]      = (__bf16)c0[i];
    r[4 + i]  = (__bf16)c1[i];
    r[8 + i]  = (__bf16)c2[i];
    r[12 + i] = (__bf16)c3[i];
  }
  return r;
}

// ---------------------------------------------------------------------------
// GEMM: Y = X (MxK) * W^T (KxN) + bias, M=8192, N=K=1024.
// Each wave: 64 rows x 16 cols (4 accumulator tiles reuse one B fragment).
// OUT_MODE 0: bf16, per-head [B,H,S,HD] (Q/K projections; scale after bias)
// OUT_MODE 1: bf16, per-head transposed [B,H,HD,S] (V projection)
// OUT_MODE 2: f32 flat [M,N] (output projection, non-temporal stores)
// ---------------------------------------------------------------------------
template <int OUT_MODE, bool XBF16>
__global__ __launch_bounds__(256) void mha_gemm(const void* __restrict__ Xv,
                                                const float* __restrict__ W,
                                                const float* __restrict__ bias,
                                                void* __restrict__ Yv,
                                                float scale) {
  const int lane  = threadIdx.x & 31;
  const int wave  = threadIdx.x >> 5;
  const int task  = blockIdx.x * 8 + wave;   // 8192 wave-tasks
  const int ntile = task & 63;               // 64 column tiles
  const int mblk  = task >> 6;               // 128 row blocks of 64
  const int nbase = ntile * 16;
  const int mbase = mblk * 64;
  const int ncol  = nbase + (lane & 15);
  const int mrow  = lane & 15;
  const int half  = lane >> 4;

  const float* Wcol = W + (size_t)ncol * ND;
  v8f acc[4] = {};

  for (int kb = 0; kb < ND; kb += 32) {
    if (kb + 32 < ND) __builtin_prefetch(Wcol + kb + 32, 0, 0);
    v16bf bfrag = load_b_f32(Wcol, kb, lane);
#pragma unroll
    for (int i = 0; i < 4; ++i) {
      const int row = mbase + 16 * i + mrow;
      v16bf afrag;
      if (XBF16)
        afrag = load_a_bf16((const __bf16*)Xv + (size_t)row * ND, kb, lane);
      else
        afrag = load_a_f32((const float*)Xv + (size_t)row * ND, kb, lane, 1.0f);
      acc[i] = wmma_bf16(afrag, bfrag, acc[i]);
    }
  }

  const float bn = bias[ncol];
#pragma unroll
  for (int i = 0; i < 4; ++i) {
#pragma unroll
    for (int e = 0; e < 8; ++e) {
      const int m = mbase + 16 * i + half * 8 + e;  // D layout: m = vgpr+8*half
      const float val = (acc[i][e] + bn) * scale;
      if (OUT_MODE == 2) {
        __builtin_nontemporal_store(val, (float*)Yv + (size_t)m * ND + ncol);
      } else {
        const int b = m >> 10, s = m & (NS - 1);
        const int h = ncol >> 6, hd = ncol & (NHD - 1);
        const size_t idx =
            (OUT_MODE == 0)
                ? ((size_t)((b * NH + h) * NS + s)) * NHD + hd
                : ((size_t)((b * NH + h) * NHD + hd)) * NS + s;
        ((__bf16*)Yv)[idx] = (__bf16)val;
      }
    }
  }
}

// ---------------------------------------------------------------------------
// Scores: per (b, h, 16-row q block): scores = Qh * Kh^T (K=64, two WMMAs),
// mask -> -inf, sigmoid (v_exp + v_rcp, no IEEE divide), stage f32 tile in
// LDS, row-sum via ds_add_f32, normalize with precomputed reciprocal, write
// attention/rowsum to d_out (NT stores), row-sums to workspace.
// ---------------------------------------------------------------------------
__global__ __launch_bounds__(256) void scores_kernel(
    const __bf16* __restrict__ Qh, const __bf16* __restrict__ Kh,
    const unsigned char* __restrict__ mask, float* __restrict__ att_norm,
    float* __restrict__ rowsum_ws) {
  __shared__ float att[16][NS];   // 64 KB (WGP has 320 KB LDS)
  __shared__ float sums[16];
  __shared__ float invs[16];

  const int b = blockIdx.z, h = blockIdx.y, qb = blockIdx.x;
  const int tid = threadIdx.x;
  const int lane = tid & 31, wave = tid >> 5, half = lane >> 4;

  if (tid < 16) sums[tid] = 0.0f;

  const __bf16* Qbase = Qh + ((size_t)(b * NH + h)) * NS * NHD;
  const __bf16* Kbase = Kh + ((size_t)(b * NH + h)) * NS * NHD;
  const unsigned char* mrow = mask + b * NS;

  const int qrow = qb * 16 + (lane & 15);
  const __bf16* Qrow = Qbase + (size_t)qrow * NHD;
  v16bf a0 = load_a_bf16(Qrow, 0, lane);    // d = 0..31
  v16bf a1 = load_a_bf16(Qrow, 32, lane);   // d = 32..63

  for (int t = 0; t < 8; ++t) {             // 8 column tiles per wave
    const int kcol = (wave * 8 + t) * 16 + (lane & 15);
    const __bf16* Kcol = Kbase + (size_t)kcol * NHD;
    v16bf b0 = load_b_bf16(Kcol, 0, lane);
    v16bf b1 = load_b_bf16(Kcol, 32, lane);
    v8f c = {};
    c = wmma_bf16(a0, b0, c);
    c = wmma_bf16(a1, b1, c);
    const bool mk = mrow[kcol] != 0;
#pragma unroll
    for (int e = 0; e < 8; ++e) {
      const float x = mk ? -__builtin_inff() : c[e];
      // sigmoid via hardware transcendentals: v_exp_f32 + v_rcp_f32.
      // sigmoid(-inf): exp(+inf)=inf -> rcp(inf)=0, matching the reference.
      att[half * 8 + e][kcol] = __builtin_amdgcn_rcpf(1.0f + __expf(-x));
    }
  }
  __syncthreads();

  {  // row sums: 16 threads per row, 64 elements each, LDS float atomics
    const int r = tid >> 4, seg = tid & 15;
    float s = 0.0f;
#pragma unroll
    for (int cc = 0; cc < 64; ++cc) s += att[r][seg * 64 + cc];
    atomicAdd(&sums[r], s);
  }
  __syncthreads();

  if (tid < 16) {
    const float s = sums[tid];
    invs[tid] = __builtin_amdgcn_rcpf(s);   // one v_rcp_f32 per row
    rowsum_ws[(b * NH + h) * NS + qb * 16 + tid] = s;
  }
  __syncthreads();

  float* outp = att_norm + (((size_t)(b * NH + h)) * NS + qb * 16) * NS;
  for (int idx = tid; idx < 16 * NS; idx += 256) {
    const int r = idx >> 10, cc = idx & (NS - 1);
    __builtin_nontemporal_store(att[r][cc] * invs[r],
                                outp + (size_t)r * NS + cc);
  }
}

// ---------------------------------------------------------------------------
// AV: context = attention * Vh.  Unnormalized attention is reconstructed as
// att_norm * rowsum[row] (per-lane scalar) while converting f32->bf16.
// B fragments come K-contiguous from VhT [B,H,HD,S].  Output ctx bf16 [B,S,D].
// ---------------------------------------------------------------------------
__global__ __launch_bounds__(256) void av_kernel(
    const float* __restrict__ att_norm, const float* __restrict__ rowsum,
    const __bf16* __restrict__ VhT, __bf16* __restrict__ ctx) {
  const int b = blockIdx.z, h = blockIdx.y;
  const int lane = threadIdx.x & 31, wave = threadIdx.x >> 5;
  const int half = lane >> 4;
  const int task = blockIdx.x * 8 + wave;   // 64 tasks per (b,h)
  const int hdtile = task & 3;              // 4 tiles cover HD=64
  const int qblk = task >> 2;               // 16 blocks of 64 q rows

  const float* A = att_norm + ((size_t)(b * NH + h)) * NS * NS;
  const float* rs = rowsum + (b * NH + h) * NS;
  const int hdcol = hdtile * 16 + (lane & 15);
  const __bf16* Vcol =
      VhT + (((size_t)(b * NH + h)) * NHD + hdcol) * NS;

  float scl[4];
#pragma unroll
  for (int i = 0; i < 4; ++i) scl[i] = rs[qblk * 64 + 16 * i + (lane & 15)];

  v8f acc[4] = {};
  for (int kb = 0; kb < NS; kb += 32) {
    v16bf bfrag = load_b_bf16(Vcol, kb, lane);
#pragma unroll
    for (int i = 0; i < 4; ++i) {
      const int row = qblk * 64 + 16 * i + (lane & 15);
      v16bf afrag = load_a_f32(A + (size_t)row * NS, kb, lane, scl[i]);
      acc[i] = wmma_bf16(afrag, bfrag, acc[i]);
    }
  }

#pragma unroll
  for (int i = 0; i < 4; ++i) {
#pragma unroll
    for (int e = 0; e < 8; ++e) {
      const int q = qblk * 64 + 16 * i + half * 8 + e;
      ctx[((size_t)(b * NS + q)) * ND + h * NHD + hdcol] = (__bf16)acc[i][e];
    }
  }
}

// ---------------------------------------------------------------------------
// Launch: 3 projections -> scores/sigmoid/normalize -> AV -> output GEMM.
// Inputs: k,v,q,mask,Wq,bq,Wk,bk,Wv,bv,Wo,bo.
// d_out = [output (8M f32) | attention_normalized (128M f32)].
// Workspace: Qh(16MB) Kh(16MB) VhT(16MB) ctx(16MB) rowsum(0.5MB) ~ 67MB.
// ---------------------------------------------------------------------------
extern "C" void kernel_launch(void* const* d_in, const int* in_sizes, int n_in,
                              void* d_out, int out_size, void* d_ws,
                              size_t ws_size, hipStream_t stream) {
  const float* k  = (const float*)d_in[0];
  const float* v  = (const float*)d_in[1];
  const float* q  = (const float*)d_in[2];
  const unsigned char* mask = (const unsigned char*)d_in[3];
  const float* Wq = (const float*)d_in[4];
  const float* bq = (const float*)d_in[5];
  const float* Wk = (const float*)d_in[6];
  const float* bk = (const float*)d_in[7];
  const float* Wv = (const float*)d_in[8];
  const float* bv = (const float*)d_in[9];
  const float* Wo = (const float*)d_in[10];
  const float* bo = (const float*)d_in[11];

  float* out = (float*)d_out;
  float* att_norm = out + (size_t)NB * NS * ND;  // second output, [B,H,S,S]

  char* ws = (char*)d_ws;
  __bf16* Qh  = (__bf16*)(ws);
  __bf16* Kh  = (__bf16*)(ws + ((size_t)16 << 20));
  __bf16* VhT = (__bf16*)(ws + ((size_t)32 << 20));
  __bf16* ctx = (__bf16*)(ws + ((size_t)48 << 20));
  float* rowsum = (float*)(ws + ((size_t)64 << 20));

  // QKV projections (Q scaled by 1/sqrt(64) after bias, like the reference).
  mha_gemm<0, false><<<1024, 256, 0, stream>>>(q, Wq, bq, Qh, 0.125f);
  mha_gemm<0, false><<<1024, 256, 0, stream>>>(k, Wk, bk, Kh, 1.0f);
  mha_gemm<1, false><<<1024, 256, 0, stream>>>(v, Wv, bv, VhT, 1.0f);

  // Scores + sigmoid + normalization (writes attention_normalized output).
  scores_kernel<<<dim3(64, NH, NB), 256, 0, stream>>>(Qh, Kh, mask, att_norm,
                                                      rowsum);

  // AV with unnormalized attention recovered as att_norm * rowsum.
  av_kernel<<<dim3(8, NH, NB), 256, 0, stream>>>(att_norm, rowsum, VhT, ctx);

  // Output projection (writes first output).
  mha_gemm<2, true><<<1024, 256, 0, stream>>>(ctx, Wo, bo, out, 1.0f);
}
